// RobustSupplyChainSAGE_18597208392307
// MI455X (gfx1250) — matrix operand order repaired and checked by reading, not compiled, not run on hardware
//
#include <hip/hip_runtime.h>

#define N_NODES 100000
#define N_EDGES 1600000
#define N_QUERY 100000
#define NODE_IN 64
#define EDGE_IN 16
#define HID 128
#define EPS 1e-5f

typedef __attribute__((ext_vector_type(16))) _Float16 v16h;
typedef __attribute__((ext_vector_type(8)))  float    v8f;

// ---------------------------------------------------------------------------
// float atomic max via sign-split int/uint atomics (native L2 atomics, no CAS)
// ---------------------------------------------------------------------------
__device__ __forceinline__ void atomic_max_f32(float* addr, float val) {
    if (val >= 0.0f) atomicMax((int*)addr, __float_as_int(val));
    else             atomicMin((unsigned int*)addr, __float_as_uint(val));
}

// ---------------------------------------------------------------------------
// in-degree count
// ---------------------------------------------------------------------------
__global__ void k_degree(const int* __restrict__ dst, float* __restrict__ cnt, int E) {
    int e = blockIdx.x * blockDim.x + threadIdx.x;
    if (e < E) atomicAdd(&cnt[dst[e]], 1.0f);
}

// ---------------------------------------------------------------------------
// init aggr buffer: [node, 0:128] = sum accum (0), [node, 128:256] = max (-inf)
// ---------------------------------------------------------------------------
__global__ void k_init_aggr(float* __restrict__ aggr, int n) {  // n = N_NODES*HID
    int i = blockIdx.x * blockDim.x + threadIdx.x;
    if (i >= n) return;
    int node = i >> 7, f = i & 127;
    aggr[(size_t)node * 256 + f]       = 0.0f;
    aggr[(size_t)node * 256 + 128 + f] = -3.402823466e38f;
}

// ---------------------------------------------------------------------------
// edge scatter: thread = (edge, 4-feature chunk); float4 gather of h[src],
// atomic add into sum half, atomic max into max half. aggr+h are L2-resident.
// ---------------------------------------------------------------------------
__global__ void k_scatter(const float* __restrict__ h, const int* __restrict__ src,
                          const int* __restrict__ dst, float* __restrict__ aggr, int E) {
    long long t = (long long)blockIdx.x * blockDim.x + threadIdx.x;
    int e = (int)(t >> 5);
    if (e >= E) return;
    int f4 = ((int)t & 31) << 2;
    int s = src[e], d = dst[e];
    const float4 v = *(const float4*)(h + (size_t)s * HID + f4);
    float* sb = aggr + (size_t)d * 256 + f4;
    float* mb = sb + 128;
    atomicAdd(sb + 0, v.x); atomicAdd(sb + 1, v.y);
    atomicAdd(sb + 2, v.z); atomicAdd(sb + 3, v.w);
    atomic_max_f32(mb + 0, v.x); atomic_max_f32(mb + 1, v.y);
    atomic_max_f32(mb + 2, v.z); atomic_max_f32(mb + 3, v.w);
}

// ---------------------------------------------------------------------------
// finalize: mean = sum / max(cnt,1); max -> 0 where cnt==0
// ---------------------------------------------------------------------------
__global__ void k_finalize(float* __restrict__ aggr, const float* __restrict__ cnt, int n) {
    int i = blockIdx.x * blockDim.x + threadIdx.x;
    if (i >= n) return;
    int node = i >> 7, f = i & 127;
    float c = cnt[node];
    size_t base = (size_t)node * 256 + f;
    aggr[base] = aggr[base] / fmaxf(c, 1.0f);
    float m = aggr[base + 128];
    aggr[base + 128] = (c > 0.0f) ? m : 0.0f;
}

// ---------------------------------------------------------------------------
// Weight repack: f32 [K,N] row-major -> f16 fragment order [tileN][chunk][lane][16]
// matching the WMMA B layout (lane = col (lane&15), j<8: K = 8*(lane>>4)+j,
// j>=8: K = 16+8*(lane>>4)+(j-8)). K-tail padded with zeros.
// cBase allows packing two matrices ([Wl;Wr]) into one chunk sequence.
// ---------------------------------------------------------------------------
__global__ void k_repack(const float* __restrict__ B, _Float16* __restrict__ out,
                         int K, int N, int Ctot, int cBase, int Cloc) {
    int idx = blockIdx.x * blockDim.x + threadIdx.x;
    int total = (N >> 4) * Cloc * 512;
    if (idx >= total) return;
    int j    = idx & 15;
    int lane = (idx >> 4) & 31;
    int rest = idx >> 9;
    int c    = rest % Cloc;
    int tn   = rest / Cloc;
    int k    = c * 32 + ((lane >> 4) << 3) + (j & 7) + ((j >> 3) << 4);
    int col  = tn * 16 + (lane & 15);
    float val = (k < K) ? B[(size_t)k * N + col] : 0.0f;
    out[(((size_t)tn * Ctot + cBase + c) << 9) + (lane << 4) + j] = (_Float16)val;
}

// ---------------------------------------------------------------------------
// WMMA GEMM:  C[M,TN] = act(scale*(A @ B + bias) + shift)
// MODE 0: A = A0 row-major, lda = TK
// MODE 1: A row = concat(h[qs[row]](128) || h[qt[row]](128) || eat[row](16))
// MODE 2: A row = concat(A0[row](256) || A1[row](128))      (aggr || h)
// One wave computes NT adjacent 16x16 tiles (A fragment reused NT times).
// All K chunks fully unrolled; A loads are unconditional b128s; B fragments
// come pre-packed in f16 (2 x b128 per fragment).
// ---------------------------------------------------------------------------
template<int MODE, int TK, int TN, int NT>
__global__ void k_gemm(const float* __restrict__ A0, const float* __restrict__ A1,
                       const _Float16* __restrict__ PB,
                       const float* __restrict__ bias,
                       const float* __restrict__ bn_g, const float* __restrict__ bn_b,
                       const float* __restrict__ bn_m, const float* __restrict__ bn_v,
                       const int* __restrict__ qs, const int* __restrict__ qt,
                       const float* __restrict__ eat,
                       float* __restrict__ C, int M, int relu) {
    constexpr int CH     = (TK + 31) / 32;
    constexpr int GROUPS = TN / (16 * NT);

    int wave = threadIdx.x >> 5;
    int lane = threadIdx.x & 31;
    int tile = blockIdx.x * (blockDim.x >> 5) + wave;
    if (tile >= (M >> 4) * GROUPS) return;
    int tm  = tile / GROUPS;
    int grp = tile - tm * GROUPS;
    int tn0 = grp * NT;
    int r    = lane & 15;
    int half = lane >> 4;
    int row  = tm * 16 + r;

    const float* base0;
    const float* base1 = nullptr;
    const float* base2 = nullptr;
    if (MODE == 0) {
        base0 = A0 + (size_t)row * TK;
    } else if (MODE == 2) {
        base0 = A0 + (size_t)row * (2 * HID);
        base1 = A1 + (size_t)row * HID;
    } else {
        base0 = A0 + (size_t)qs[row] * HID;
        base1 = A0 + (size_t)qt[row] * HID;
        base2 = eat + (size_t)row * EDGE_IN;
    }

    v8f acc[NT];
#pragma unroll
    for (int nt = 0; nt < NT; ++nt)
#pragma unroll
        for (int i = 0; i < 8; ++i) acc[nt][i] = 0.0f;

#pragma unroll
    for (int c = 0; c < CH; ++c) {
        const int kb = c * 32;
        // chunk source pointer (32 is a divisor of every segment boundary)
        const float* p;
        if (MODE == 0)      p = base0 + kb;
        else if (MODE == 2) p = (kb < 2 * HID) ? base0 + kb : base1 + (kb - 2 * HID);
        else                p = (kb < HID)     ? base0 + kb
                              : (kb < 2 * HID) ? base1 + (kb - HID)
                                               : base2 + (kb - 2 * HID);
        // A fragment: two contiguous 8-float runs per lane
        const float4 q0 = *(const float4*)(p + 8 * half);
        const float4 q1 = *(const float4*)(p + 8 * half + 4);
        float4 q2, q3;
        if (kb + 16 < TK) {  // constant-folds after unroll; false only for K%32==16 tail
            q2 = *(const float4*)(p + 16 + 8 * half);
            q3 = *(const float4*)(p + 20 + 8 * half);
        } else {
            q2 = float4{0.0f, 0.0f, 0.0f, 0.0f};
            q3 = float4{0.0f, 0.0f, 0.0f, 0.0f};
        }
        v16h a;
        a[0]  = (_Float16)q0.x; a[1]  = (_Float16)q0.y;
        a[2]  = (_Float16)q0.z; a[3]  = (_Float16)q0.w;
        a[4]  = (_Float16)q1.x; a[5]  = (_Float16)q1.y;
        a[6]  = (_Float16)q1.z; a[7]  = (_Float16)q1.w;
        a[8]  = (_Float16)q2.x; a[9]  = (_Float16)q2.y;
        a[10] = (_Float16)q2.z; a[11] = (_Float16)q2.w;
        a[12] = (_Float16)q3.x; a[13] = (_Float16)q3.y;
        a[14] = (_Float16)q3.z; a[15] = (_Float16)q3.w;

#pragma unroll
        for (int nt = 0; nt < NT; ++nt) {
            const v16h b = *(const v16h*)(PB + ((((size_t)(tn0 + nt) * CH + c) << 9)
                                               + (lane << 4)));
            acc[nt] = __builtin_amdgcn_wmma_f32_16x16x32_f16(
                false, a, false, b, (short)0, acc[nt], false, false);
        }
    }

#pragma unroll
    for (int nt = 0; nt < NT; ++nt) {
        int col = (tn0 + nt) * 16 + r;
        float bc = bias ? bias[col] : 0.0f;
        float sc = 1.0f, sh = 0.0f;
        if (bn_g) {
            float inv = rsqrtf(bn_v[col] + EPS);
            sc = bn_g[col] * inv;
            sh = bn_b[col] - bn_m[col] * sc;
        }
#pragma unroll
        for (int i = 0; i < 8; ++i) {
            float y = acc[nt][i] + bc;
            y = y * sc + sh;
            if (relu) y = fmaxf(y, 0.0f);
            C[(size_t)(tm * 16 + i + 8 * half) * TN + col] = y;
        }
    }
}

// ---------------------------------------------------------------------------
// fused LayerNorm + ReLU + residual: h = relu(LN(t)*g+b) + h  (wave32-per-row)
// ---------------------------------------------------------------------------
__global__ void k_ln_relu_res(const float* __restrict__ t, const float* __restrict__ g,
                              const float* __restrict__ b, float* __restrict__ h, int Nn) {
    int wave = threadIdx.x >> 5, lane = threadIdx.x & 31;
    int row = blockIdx.x * 8 + wave;
    if (row >= Nn) return;
    const float* tr = t + (size_t)row * HID;
    float v[4], s = 0.0f, ss = 0.0f;
#pragma unroll
    for (int i = 0; i < 4; ++i) {
        v[i] = tr[lane + 32 * i];
        s += v[i];
        ss += v[i] * v[i];
    }
#pragma unroll
    for (int m = 16; m >= 1; m >>= 1) {
        s  += __shfl_xor(s,  m, 32);
        ss += __shfl_xor(ss, m, 32);
    }
    float mu  = s * (1.0f / HID);
    float var = ss * (1.0f / HID) - mu * mu;
    float inv = rsqrtf(var + EPS);
    float* hr = h + (size_t)row * HID;
#pragma unroll
    for (int i = 0; i < 4; ++i) {
        int f = lane + 32 * i;
        float y = (v[i] - mu) * inv * g[f] + b[f];
        y = fmaxf(y, 0.0f);
        hr[f] = y + hr[f];
    }
}

// ---------------------------------------------------------------------------
// final head: out[m] = t1[m,:] . W3 + b3   (N=1, wave-per-row reduction)
// ---------------------------------------------------------------------------
__global__ void k_head(const float* __restrict__ t1, const float* __restrict__ W3,
                       const float* __restrict__ b3, float* __restrict__ out, int M) {
    int wave = threadIdx.x >> 5, lane = threadIdx.x & 31;
    int row = blockIdx.x * 8 + wave;
    if (row >= M) return;
    const float* tr = t1 + (size_t)row * HID;
    float acc = 0.0f;
#pragma unroll
    for (int i = 0; i < 4; ++i) {
        int f = lane + 32 * i;
        acc += tr[f] * W3[f];
    }
#pragma unroll
    for (int m = 16; m >= 1; m >>= 1) acc += __shfl_xor(acc, m, 32);
    if (lane == 0) out[row] = acc + b3[0];
}

// ---------------------------------------------------------------------------
extern "C" void kernel_launch(void* const* d_in, const int* in_sizes, int n_in,
                              void* d_out, int out_size, void* d_ws, size_t ws_size,
                              hipStream_t stream) {
    (void)in_sizes; (void)n_in; (void)out_size; (void)ws_size;

    const float* x    = (const float*)d_in[0];
    const int*   ei   = (const int*)d_in[1];
    const float* eat  = (const float*)d_in[2];
    const int*   qei  = (const int*)d_in[3];
    const float* encW = (const float*)d_in[4];
    const float* encB = (const float*)d_in[5];
    const float *Wl[3], *bl[3], *Wr[3];
    for (int l = 0; l < 3; ++l) {
        Wl[l] = (const float*)d_in[6 + 3 * l];
        bl[l] = (const float*)d_in[7 + 3 * l];
        Wr[l] = (const float*)d_in[8 + 3 * l];
    }
    const float *ng[3], *nb[3];
    for (int l = 0; l < 3; ++l) {
        ng[l] = (const float*)d_in[15 + 2 * l];
        nb[l] = (const float*)d_in[16 + 2 * l];
    }
    const float* W1  = (const float*)d_in[21];
    const float* b1  = (const float*)d_in[22];
    const float* bng = (const float*)d_in[23];
    const float* bnb = (const float*)d_in[24];
    const float* bnm = (const float*)d_in[25];
    const float* bnv = (const float*)d_in[26];
    const float* W2  = (const float*)d_in[27];
    const float* b2  = (const float*)d_in[28];
    const float* W3  = (const float*)d_in[29];
    const float* b3  = (const float*)d_in[30];

    const int* src = ei;
    const int* dst = ei + N_EDGES;
    const int* qs  = qei;
    const int* qt  = qei + N_QUERY;

    // workspace layout
    char* ws = (char*)d_ws;
    size_t off = 0;
    auto walloc = [&](size_t bytes) -> void* {
        void* p = ws + off;
        off = (off + bytes + 255) & ~(size_t)255;
        return p;
    };
    float* cnt = (float*)walloc((size_t)N_NODES * 4);
    float* h   = (float*)walloc((size_t)N_NODES * HID * 4);
    float* t1  = (float*)walloc((size_t)N_NODES * HID * 4);
    float* t0  = (float*)walloc((size_t)N_NODES * 256 * 4);
    // packed f16 weights (fragment order)
    _Float16* pEnc  = (_Float16*)walloc((size_t)8  * 2  * 512 * 2);
    _Float16* pL[3];
    for (int l = 0; l < 3; ++l)
        pL[l]       = (_Float16*)walloc((size_t)8  * 12 * 512 * 2);
    _Float16* pW1   = (_Float16*)walloc((size_t)16 * 9  * 512 * 2);
    _Float16* pW2   = (_Float16*)walloc((size_t)8  * 8  * 512 * 2);

    // ---- weight repack (tiny; runs each call) ----
    auto repack = [&](const float* B, _Float16* outp, int K, int N,
                      int Ctot, int cBase, int Cloc) {
        int total = (N >> 4) * Cloc * 512;
        k_repack<<<(total + 255) / 256, 256, 0, stream>>>(B, outp, K, N, Ctot, cBase, Cloc);
    };
    repack(encW, pEnc, NODE_IN, HID, 2, 0, 2);
    for (int l = 0; l < 3; ++l) {
        repack(Wl[l], pL[l], 2 * HID, HID, 12, 0, 8);   // chunks 0..7  : Wl (K=256)
        repack(Wr[l], pL[l], HID,     HID, 12, 8, 4);   // chunks 8..11 : Wr (K=128)
    }
    repack(W1, pW1, 2 * HID + EDGE_IN, 2 * HID, 9, 0, 9);
    repack(W2, pW2, 2 * HID,           HID,     8, 0, 8);

    // ---- degree ----
    hipMemsetAsync(cnt, 0, (size_t)N_NODES * sizeof(float), stream);
    k_degree<<<(N_EDGES + 255) / 256, 256, 0, stream>>>(dst, cnt, N_EDGES);

    // ---- encoder: h = x @ enc_W + enc_b ----
    {
        int waves = (N_NODES / 16) * (HID / 64);   // GROUPS = 128/(16*4) = 2
        k_gemm<0, NODE_IN, HID, 4><<<(waves + 3) / 4, 128, 0, stream>>>(
            x, nullptr, pEnc, encB, nullptr, nullptr, nullptr, nullptr,
            nullptr, nullptr, nullptr, h, N_NODES, 0);
    }

    const int nF = N_NODES * HID;
    for (int l = 0; l < 3; ++l) {
        k_init_aggr<<<(nF + 255) / 256, 256, 0, stream>>>(t0, nF);
        long long sthreads = (long long)N_EDGES * 32;
        k_scatter<<<(unsigned)((sthreads + 255) / 256), 256, 0, stream>>>(h, src, dst, t0, N_EDGES);
        k_finalize<<<(nF + 255) / 256, 256, 0, stream>>>(t0, cnt, nF);
        // t1 = [mean||max] @ Wl + h @ Wr + bl   (fused K=384 GEMM)
        int waves = (N_NODES / 16) * 2;
        k_gemm<2, 3 * HID, HID, 4><<<(waves + 3) / 4, 128, 0, stream>>>(
            t0, h, pL[l], bl[l], nullptr, nullptr, nullptr, nullptr,
            nullptr, nullptr, nullptr, t1, N_NODES, 0);
        // h = relu(LN(t1)) + h
        k_ln_relu_res<<<(N_NODES + 7) / 8, 256, 0, stream>>>(t1, ng[l], nb[l], h, N_NODES);
    }

    // ---- MLP stage 1 (fused concat gather + BN + relu), K = 272 ----
    {
        int waves = (N_QUERY / 16) * 4;            // GROUPS = 256/(16*4) = 4
        k_gemm<1, 2 * HID + EDGE_IN, 2 * HID, 4><<<(waves + 3) / 4, 128, 0, stream>>>(
            h, nullptr, pW1, b1, bng, bnb, bnm, bnv,
            qs, qt, eat, t0, N_QUERY, 1);
    }
    // ---- MLP stage 2: t1 = relu(t0 @ W2 + b2) ----
    {
        int waves = (N_QUERY / 16) * 2;
        k_gemm<0, 2 * HID, HID, 4><<<(waves + 3) / 4, 128, 0, stream>>>(
            t0, nullptr, pW2, b2, nullptr, nullptr, nullptr, nullptr,
            nullptr, nullptr, nullptr, t1, N_QUERY, 1);
    }
    // ---- head ----
    k_head<<<(N_QUERY + 7) / 8, 256, 0, stream>>>(t1, W3, b3, (float*)d_out, N_QUERY);
}
